// PRM_7352984011191
// MI455X (gfx1250) — matrix operand order repaired and checked by reading, not compiled, not run on hardware
//
#include <hip/hip_runtime.h>
#include <hip/hip_bf16.h>

typedef __attribute__((ext_vector_type(16))) _Float16 v16h;
typedef __attribute__((ext_vector_type(8)))  _Float16 v8h;
typedef __attribute__((ext_vector_type(4)))  _Float16 v4h;
typedef __attribute__((ext_vector_type(2)))  _Float16 v2h;
typedef __attribute__((ext_vector_type(8)))  float    v8f;

#define NB       4
#define CD       64
#define HWDIM    4096
#define STILE    64
#define TTILE    64
#define NWAVE    4
#define NTHREADS 128
#define QS       72   // LDS row strides in halves (144B, 16B-aligned)
#define KS       72
#define VS       72

union V8   { v8f  v; float    f[8]; };
union V16H { v16h v; v2h      h2[8]; };
union V4H  { v4h  v; v2h      h2[2]; };

__device__ __forceinline__ v16h cat16(v8h a, v8h b) {
    return __builtin_shufflevector(a, b, 0,1,2,3,4,5,6,7,8,9,10,11,12,13,14,15);
}

// packed f32 -> 2 x f16 convert (v_cvt_pkrtz_f16_f32)
__device__ __forceinline__ v2h pk2(float a, float b) {
    auto p = __builtin_amdgcn_cvt_pkrtz(a, b);
    v2h r;
    __builtin_memcpy(&r, &p, sizeof(r));
    return r;
}

// Each thread owns two 4x4 sub-blocks of the 64(c) x 64(t) tile:
// block (bc0 + 8*blk, bt).  Rows are float4 loads along t (coalesced:
// 16 lanes x 16B = 256B per c-row segment).
__device__ __forceinline__ void load_tile_regs(const float* __restrict__ lpn,
                                               int t0, int bt, int bc0, float4* st) {
#pragma unroll
    for (int blk = 0; blk < 2; ++blk) {
        const int cb = (bc0 + 8 * blk) * 4;
#pragma unroll
        for (int r = 0; r < 4; ++r)
            st[blk * 4 + r] = *(const float4*)(lpn + (cb + r) * HWDIM + t0 + bt * 4);
    }
}

// Convert with packed cvt_pkrtz, emit packed 8B stores for BOTH layouts:
//   V[c][t]: row r of the 4x4 block  -> ds_store_b64 along t
//   K[t][c]: column k of the block   -> ds_store_b64 along c
__device__ __forceinline__ void store_tile(const float4* st, int bt, int bc0,
                                           _Float16* __restrict__ K,
                                           _Float16* __restrict__ V) {
#pragma unroll
    for (int blk = 0; blk < 2; ++blk) {
        const int cb = (bc0 + 8 * blk) * 4;
        const int tb = bt * 4;
        float a[4][4];
#pragma unroll
        for (int r = 0; r < 4; ++r) {
            const float4 v = st[blk * 4 + r];
            a[r][0] = v.x; a[r][1] = v.y; a[r][2] = v.z; a[r][3] = v.w;
        }
#pragma unroll
        for (int r = 0; r < 4; ++r) {
            V4H vr;
            vr.h2[0] = pk2(a[r][0], a[r][1]);
            vr.h2[1] = pk2(a[r][2], a[r][3]);
            *(v4h*)&V[(cb + r) * VS + tb] = vr.v;
        }
#pragma unroll
        for (int k = 0; k < 4; ++k) {
            V4H kc;
            kc.h2[0] = pk2(a[0][k], a[1][k]);
            kc.h2[1] = pk2(a[2][k], a[3][k]);
            *(v4h*)&K[(tb + k) * KS + cb] = kc.v;
        }
    }
}

__global__ __launch_bounds__(NTHREADS)
void flash_attn_f16_wmma(const float* __restrict__ lp,   // K and V: [N][C][HW]
                         const float* __restrict__ ulp,  // Q source: [N][C][HW]
                         float* __restrict__ out)        // [N][C][HW]
{
    __shared__ alignas(16) _Float16 Qlds[STILE * QS];          // [s][c]
    __shared__ alignas(16) _Float16 Klds[2][TTILE * KS];       // double-buffered [t][c]
    __shared__ alignas(16) _Float16 Vlds[2][CD * VS];          // double-buffered [c][t]

    const int tid  = threadIdx.x;
    const int wave = tid >> 5;
    const int lane = tid & 31;
    const int l16  = lane & 15;
    const int hi   = lane >> 4;          // 0: lanes 0-15, 1: lanes 16-31
    const int bt   = tid & 15;           // staging: t-block
    const int bc0  = tid >> 4;           // staging: first c-block

    const int n  = blockIdx.y;
    const int s0 = blockIdx.x * STILE;

    const float* lpn  = lp  + (size_t)n * CD * HWDIM;
    const float* ulpn = ulp + (size_t)n * CD * HWDIM;

    // ---------- stage Q tile: global [c][s] (f32) -> LDS [s][c] (f16) ----------
#pragma unroll
    for (int i = 0; i < 8; ++i) {
        int idx4 = i * NTHREADS + tid;
        int c    = idx4 >> 4;
        int s4   = (idx4 & 15) << 2;
        const float4 v = *(const float4*)(ulpn + c * HWDIM + s0 + s4);
        Qlds[(s4 + 0) * QS + c] = (_Float16)v.x;
        Qlds[(s4 + 1) * QS + c] = (_Float16)v.y;
        Qlds[(s4 + 2) * QS + c] = (_Float16)v.z;
        Qlds[(s4 + 3) * QS + c] = (_Float16)v.w;
    }

    // ---------- prologue of the K/V software pipeline ----------
    float4 stage[8];
    load_tile_regs(lpn, 0, bt, bc0, stage);

    __syncthreads();   // Q tile ready

    // ---------- Q B-fragments (lane = column s, K-dim = c), loaded once ----------
    const int sw = wave * 16;            // this wave's 16 softmax columns
    v16h qb[2];
#pragma unroll
    for (int cc = 0; cc < 2; ++cc) {
        const int c0 = cc * 32 + hi * 16;
        v8h b0 = *(const v8h*)&Qlds[(sw + l16) * QS + c0];
        v8h b1 = *(const v8h*)&Qlds[(sw + l16) * QS + c0 + 8];
        qb[cc] = cat16(b0, b1);
    }

    // constant all-ones B fragment: one extra WMMA accumulates softmax row sums
    const v16h onesh = (v16h)(_Float16)1.0f;

    // ---------- flash-attention state ----------
    float mstate = -3.0e38f;             // per-lane: one softmax column
    V8 o[4], lsum;
#pragma unroll
    for (int ct = 0; ct < 4; ++ct) o[ct].v = (v8f)0.0f;
    lsum.v = (v8f)0.0f;

    int buf = 0;

    for (int t0 = 0; t0 < HWDIM; t0 += TTILE) {
        // commit prefetched tile to this iteration's LDS buffer
        store_tile(stage, bt, bc0, Klds[buf], Vlds[buf]);
        __syncthreads();                       // one barrier per iteration

        // prefetch next tile: overlaps all compute below
        if (t0 + TTILE < HWDIM)
            load_tile_regs(lpn, t0 + TTILE, bt, bc0, stage);

        // ---- transposed energy: E^T[64t x 16s] = K(64x64) x Q^T ----
        // A = K rows (t), B = Q columns (s).  Lane holds column s = sw+l16,
        // with t values: acc[ta], VGPR j  <->  t = 16*ta + j + 8*hi.
        V8 acc[4];
#pragma unroll
        for (int ta = 0; ta < 4; ++ta) {
            acc[ta].v = (v8f)0.0f;
            const int trow = ta * 16 + l16;
#pragma unroll
            for (int cc = 0; cc < 2; ++cc) {
                const int c0 = cc * 32 + hi * 16;
                v8h lo = *(const v8h*)&Klds[buf][trow * KS + c0];
                v8h hh = *(const v8h*)&Klds[buf][trow * KS + c0 + 8];
                v16h af = cat16(lo, hh);
                acc[ta].v = __builtin_amdgcn_wmma_f32_16x16x32_f16(
                    false, af, false, qb[cc], (short)0, acc[ta].v, false, false);
            }
        }

        // ---- online softmax: max over t is in-lane + one xor-16 shuffle ----
        float mt = acc[0].f[0];
#pragma unroll
        for (int ta = 0; ta < 4; ++ta)
#pragma unroll
            for (int j = 0; j < 8; ++j)
                mt = fmaxf(mt, acc[ta].f[j]);
        mt = fmaxf(mt, __shfl_xor(mt, 16, 32));
        const float mnew  = fmaxf(mstate, mt);
        const float scale = __expf(mstate - mnew);
        mstate = mnew;

        // ---- P = exp(E - m): packed f32 subtract, exp, packed f16 convert.
        // pf[tc] is already the A-fragment layout — no LDS bounce.
        const v8f mnewv = (v8f)mnew;
        v16h pf[2];
#pragma unroll
        for (int tc = 0; tc < 2; ++tc) {
            V8 e0, e1;
            e0.v = acc[2 * tc].v     - mnewv;   // v_pk_add_f32 candidates
            e1.v = acc[2 * tc + 1].v - mnewv;
            V16H u;
#pragma unroll
            for (int j = 0; j < 8; j += 2) {
                u.h2[j >> 1]       = pk2(__expf(e0.f[j]), __expf(e0.f[j + 1]));
                u.h2[4 + (j >> 1)] = pk2(__expf(e1.f[j]), __expf(e1.f[j + 1]));
            }
            pf[tc] = u.v;
        }

        // ---- rescale O and lsum: broadcast scale to row layout, packed muls ----
        V8 scv;
#pragma unroll
        for (int j = 0; j < 8; ++j)
            scv.f[j] = __shfl(scale, j + 8 * hi, 32);  // scale of row s = sw+j+8hi
#pragma unroll
        for (int ct = 0; ct < 4; ++ct) o[ct].v *= scv.v;   // v_pk_mul_f32 candidates
        lsum.v *= scv.v;

        // ---- O[16s x 64c] += P x V^T ; lsum += P x 1 (denominator) ----
#pragma unroll
        for (int tc = 0; tc < 2; ++tc) {
            lsum.v = __builtin_amdgcn_wmma_f32_16x16x32_f16(
                false, pf[tc], false, onesh, (short)0, lsum.v, false, false);
            const int t0c = tc * 32 + hi * 8;
#pragma unroll
            for (int ct = 0; ct < 4; ++ct) {
                const int crow = ct * 16 + l16;   // column of O = channel c
                v8h b0 = *(const v8h*)&Vlds[buf][crow * VS + t0c];
                v8h b1 = *(const v8h*)&Vlds[buf][crow * VS + t0c + 16];
                v16h bf = cat16(b0, b1);
                o[ct].v = __builtin_amdgcn_wmma_f32_16x16x32_f16(
                    false, pf[tc], false, bf, (short)0, o[ct].v, false, false);
            }
        }
        buf ^= 1;
    }

    // ---------- normalize (packed) and store: out[n][c][s] ----------
    V8 rlv;
#pragma unroll
    for (int j = 0; j < 8; ++j) rlv.f[j] = 1.0f / lsum.f[j];
#pragma unroll
    for (int ct = 0; ct < 4; ++ct) {
        o[ct].v *= rlv.v;
        const int c = ct * 16 + l16;
        float* op = out + ((size_t)n * CD + c) * HWDIM + s0 + sw;
#pragma unroll
        for (int j = 0; j < 8; ++j)
            op[j + 8 * hi] = o[ct].f[j];
    }
}

extern "C" void kernel_launch(void* const* d_in, const int* in_sizes, int n_in,
                              void* d_out, int out_size, void* d_ws, size_t ws_size,
                              hipStream_t stream) {
    const float* lp  = (const float*)d_in[0];   // l_p  [4][64][64][64]
    const float* ulp = (const float*)d_in[1];   // ul_p [4][64][64][64]
    float* out = (float*)d_out;
    dim3 grid(HWDIM / STILE, NB);               // 64 S-tiles x 4 batches
    flash_attn_f16_wmma<<<grid, NTHREADS, 0, stream>>>(lp, ulp, out);
}
